// AFMLayer_17463337025635
// MI455X (gfx1250) — matrix operand order repaired and checked by reading, not compiled, not run on hardware
//
#include <hip/hip_runtime.h>
#include <hip/hip_bf16.h>

typedef __attribute__((ext_vector_type(16))) _Float16 v16h;
typedef __attribute__((ext_vector_type(8)))  _Float16 v8h;
typedef __attribute__((ext_vector_type(8)))  float    v8f;

#define BDIM   256
#define NWAVE  8
#define FDIM   64
#define EDIM   32
#define NPAIR  2016      // 64*63/2
#define NTILE  126       // 2016 / 16
#define XH_STR 48        // x  f16 row stride (96B, 16B aligned)
#define XT_STR 72        // x^T f16 row stride (144B, 16B aligned)
#define WT_STR 72        // W~ f16 row stride (144B, 16B aligned)

__global__ __launch_bounds__(BDIM) void afm_kernel(
    const float* __restrict__ x,
    const float* __restrict__ attn_w,   // [32,4]
    const float* __restrict__ attn_b,   // [4]
    const float* __restrict__ attn_h,   // [4,1]
    const float* __restrict__ pool_w,   // [32,1]
    const float* __restrict__ pool_b,   // [1]
    float* __restrict__ out)            // [B,1]
{
    __shared__ __align__(16) _Float16 xh[FDIM * XH_STR];   // 6144 B  x[b] f16
    __shared__ __align__(16) _Float16 xt[EDIM * XT_STR];   // 4608 B  x[b]^T f16
    __shared__ __align__(16) _Float16 wt[FDIM * WT_STR];   // 9216 B  symmetric W~ f16
    __shared__ __align__(16) float    sc[NPAIR];           // 8064 B  scores
    __shared__ unsigned short         ptab[NPAIR];         // 4032 B
    __shared__ float redA[NWAVE];
    __shared__ float redB[NWAVE];

    const int b    = blockIdx.x;
    const int tid  = threadIdx.x;
    const int lane = tid & 31;
    const int wave = tid >> 5;
    const int n    = lane & 15;
    const int c    = lane >> 4;

    // ---- stage x[b] into LDS (f16 row-major + f16 transposed), zero W~ ----
    const float4* x4 = (const float4*)(x + (size_t)b * (FDIM * EDIM));
    for (int idx = tid; idx < (FDIM * EDIM) / 4; idx += BDIM) {
        float4 v = x4[idx];
        int row = idx >> 3;            // (idx*4)/32
        int col = (idx & 7) << 2;
        _Float16* hd = &xh[row * XH_STR + col];
        _Float16 h0 = (_Float16)v.x, h1 = (_Float16)v.y;
        _Float16 h2 = (_Float16)v.z, h3 = (_Float16)v.w;
        hd[0] = h0; hd[1] = h1; hd[2] = h2; hd[3] = h3;
        xt[(col + 0) * XT_STR + row] = h0;
        xt[(col + 1) * XT_STR + row] = h1;
        xt[(col + 2) * XT_STR + row] = h2;
        xt[(col + 3) * XT_STR + row] = h3;
    }
    for (int k = tid; k < (FDIM * WT_STR) / 2; k += BDIM)
        ((unsigned int*)wt)[k] = 0u;

    // ---- pair table p -> (i,j), i<j, triu row-major order ----
    for (int p = tid; p < NPAIR; p += BDIM) {
        float q = (float)(127 * 127 - 8 * p);
        int i = (int)((127.0f - sqrtf(q)) * 0.5f);
        if (i < 0) i = 0; if (i > 62) i = 62;
        while (i < 62 && ((i + 1) * (127 - (i + 1))) / 2 <= p) ++i;
        while (i > 0 && (i * (127 - i)) / 2 > p) --i;
        int j = i + 1 + (p - (i * (127 - i)) / 2);
        ptab[p] = (unsigned short)((i << 8) | j);
    }

    // ---- constant A fragment for pass 1: attn_w^T padded to 16x32 f16 ----
    v16h afrag;
#pragma unroll
    for (int e = 0; e < 16; ++e) {
        int K = c * 16 + e;
        float w = (n < 4) ? attn_w[K * 4 + n] : 0.0f;
        afrag[e] = (_Float16)w;
    }
    float hv[4], bv[4];
#pragma unroll
    for (int v = 0; v < 4; ++v) {
        hv[v] = (c == 0) ? attn_h[v] : 0.0f;
        bv[v] = (c == 0) ? attn_b[v] : 0.0f;
    }

    __syncthreads();

    // ---- pass 1: scores via WMMA; D[M=attn unit][N=pair] -> intra-lane score ----
    for (int tile = wave; tile < NTILE; tile += NWAVE) {
        int p = tile * 16 + n;
        unsigned short pr = ptab[p];
        int i = pr >> 8, j = pr & 255;
        const _Float16* xi = &xh[i * XH_STR + c * 16];
        const _Float16* xj = &xh[j * XH_STR + c * 16];
        v8h xiL = *(const v8h*)(xi);
        v8h xiH = *(const v8h*)(xi + 8);
        v8h xjL = *(const v8h*)(xj);
        v8h xjH = *(const v8h*)(xj + 8);
        v8h pL = xiL * xjL;
        v8h pH = xiH * xjH;
        v16h bfrag;
#pragma unroll
        for (int e = 0; e < 8; ++e) { bfrag[e] = pL[e]; bfrag[e + 8] = pH[e]; }

        v8f cz = {0.f, 0.f, 0.f, 0.f, 0.f, 0.f, 0.f, 0.f};
        v8f d = __builtin_amdgcn_wmma_f32_16x16x32_f16(
            false, afrag, false, bfrag, (short)0, cz, false, false);

        float t = 0.0f;
#pragma unroll
        for (int v = 0; v < 4; ++v)
            t += hv[v] * fmaxf(d[v] + bv[v], 0.0f);
        if (c == 0) sc[p] = t;
    }
    __syncthreads();

    // ---- softmax: global max ----
    float m = -3.4e38f;
    for (int p = tid; p < NPAIR; p += BDIM) m = fmaxf(m, sc[p]);
#pragma unroll
    for (int sh = 16; sh > 0; sh >>= 1) m = fmaxf(m, __shfl_xor(m, sh, 32));
    if (lane == 0) redA[wave] = m;
    __syncthreads();
    float gmax = redA[0];
#pragma unroll
    for (int w = 1; w < NWAVE; ++w) gmax = fmaxf(gmax, redA[w]);

    // ---- exp + scatter into symmetric W~ (each thread owns its p's) ----
    float s = 0.0f;
    for (int p = tid; p < NPAIR; p += BDIM) {
        float ev = __expf(sc[p] - gmax);
        s += ev;
        unsigned short pr = ptab[p];
        int i = pr >> 8, j = pr & 255;
        _Float16 h = (_Float16)ev;
        wt[i * WT_STR + j] = h;
        wt[j * WT_STR + i] = h;
    }
#pragma unroll
    for (int sh = 16; sh > 0; sh >>= 1) s += __shfl_xor(s, sh, 32);
    if (lane == 0) redB[wave] = s;
    __syncthreads();                     // W~ complete + sums visible
    float gsum = 0.0f;
#pragma unroll
    for (int w = 0; w < NWAVE; ++w) gsum += redB[w];
    const float inv = 1.0f / gsum;

    // ---- pass 2 as WMMA GEMM: Z = X^T @ W~  (32x64, K=64), 1 tile per wave ----
    const int et = wave & 1;     // e-tile (M)
    const int it = wave >> 1;    // i-tile (N)
    v8f acc = {0.f, 0.f, 0.f, 0.f, 0.f, 0.f, 0.f, 0.f};
#pragma unroll
    for (int kt = 0; kt < 2; ++kt) {
        // A = X^T : row M = et*16+n, K chunks c*8..c*8+7 and 16+c*8..
        const _Float16* at = &xt[(et * 16 + n) * XT_STR + kt * 32 + c * 8];
        v8h aL = *(const v8h*)(at);
        v8h aH = *(const v8h*)(at + 16);
        v16h af;
#pragma unroll
        for (int e = 0; e < 8; ++e) { af[e] = aL[e]; af[e + 8] = aH[e]; }
        // B = W~ : B[K][n] = wt[it*16+n][kt*32 + K], K chunks c*8..c*8+7 and 16+c*8.. (symmetry!)
        const _Float16* bt = &wt[(it * 16 + n) * WT_STR + kt * 32 + c * 8];
        v8h bL = *(const v8h*)(bt);
        v8h bH = *(const v8h*)(bt + 16);
        v16h bf;
#pragma unroll
        for (int e = 0; e < 8; ++e) { bf[e] = bL[e]; bf[e + 8] = bH[e]; }

        acc = __builtin_amdgcn_wmma_f32_16x16x32_f16(
            false, af, false, bf, (short)0, acc, false, false);
    }

    // ---- contraction: val = sum_v X[i][e_v] * Z[e_v][i] * pw[e_v] ----
    // lane (n,c): i = it*16+n ; e_v = et*16 + 8c + v (contiguous) -> one v8h load
    {
        const _Float16* xv = &xh[(it * 16 + n) * XH_STR + et * 16 + c * 8];
        v8h xvv = *(const v8h*)(xv);
        float val = 0.0f;
#pragma unroll
        for (int v = 0; v < 8; ++v) {
            float pw = pool_w[et * 16 + c * 8 + v];
            val += (float)xvv[v] * acc[v] * pw;
        }
#pragma unroll
        for (int sh = 16; sh > 0; sh >>= 1) val += __shfl_xor(val, sh, 32);
        if (lane == 0) redA[wave] = val;
    }
    __syncthreads();
    if (tid == 0) {
        float tot = 0.0f;
#pragma unroll
        for (int w = 0; w < NWAVE; ++w) tot += redA[w];
        out[b] = 0.5f * tot * inv + pool_b[0];   // 0.5: symmetric double count
    }
}

extern "C" void kernel_launch(void* const* d_in, const int* in_sizes, int n_in,
                              void* d_out, int out_size, void* d_ws, size_t ws_size,
                              hipStream_t stream) {
    const float* x      = (const float*)d_in[0];
    const float* attn_w = (const float*)d_in[1];
    const float* attn_b = (const float*)d_in[2];
    const float* attn_h = (const float*)d_in[3];
    const float* pool_w = (const float*)d_in[4];
    const float* pool_b = (const float*)d_in[5];
    float* out = (float*)d_out;

    int B = in_sizes[0] / (FDIM * EDIM);   // 2048
    afm_kernel<<<B, BDIM, 0, stream>>>(x, attn_w, attn_b, attn_h,
                                       pool_w, pool_b, out);
}
